// MultiHeadAttention_88399016886774
// MI455X (gfx1250) — compile-verified, matrix operand order checked by reading, not run on hardware
//
#include <hip/hip_runtime.h>

typedef _Float16 v4h  __attribute__((ext_vector_type(4)));
typedef _Float16 v8h  __attribute__((ext_vector_type(8)));
typedef _Float16 v16h __attribute__((ext_vector_type(16)));
typedef float    v8f  __attribute__((ext_vector_type(8)));

#define EMB   1024
#define HEADS 16
#define HDIM  64
#define BATCH 4
#define SEQ   2048
#define ROWS  (BATCH * SEQ)   // 8192

#define S_STRIDE 36   // padded f32 score row stride (dwords) -> conflict-free b128 reads
#define P_STRIDE 40   // padded f16 P row stride (halfs), keeps 16B alignment

// ---------------------------------------------------------------------------
// 16-bit A/B WMMA fragment = two contiguous 16B runs from one row.
// lane<16 holds k{0..7,16..23}, lane>=16 holds k{8..15,24..31}; caller folds
// the +8 (hi8) into `base`.
// ---------------------------------------------------------------------------
__device__ __forceinline__ v16h ld_frag(const _Float16* __restrict__ rowp, int base) {
  v8h lo = *(const v8h*)(rowp + base);
  v8h hi = *(const v8h*)(rowp + base + 16);
  return __builtin_shufflevector(lo, hi, 0,1,2,3,4,5,6,7,8,9,10,11,12,13,14,15);
}

__device__ __forceinline__ v8f wmma_f16(v16h a, v16h b, v8f c) {
  return __builtin_amdgcn_wmma_f32_16x16x32_f16(false, a, false, b, (short)0, c,
                                                false, false);
}

// ---------------------------------------------------------------------------
// fp32 -> fp16 down-convert (vectorized x4)
// ---------------------------------------------------------------------------
__global__ void cvt_f32_f16(const float4* __restrict__ src, v4h* __restrict__ dst,
                            int n4) {
  int i = blockIdx.x * blockDim.x + threadIdx.x;
  if (i < n4) {
    float4 v = src[i];
    v4h o = { (_Float16)v.x, (_Float16)v.y, (_Float16)v.z, (_Float16)v.w };
    dst[i] = o;
  }
}

// ---------------------------------------------------------------------------
// GEMM: Y = X @ W^T + b.  X:[ROWS,EMB] f16, W:[EMB,EMB] f16 (row = out feat).
// One wave computes a 32(M) x 64(N) strip: 2 A-frags x 4 B-frags -> 8 WMMAs
// per k-step with 6 fragment loads (B reused across both M tiles).
// MODE 0: Q -> o16 = Qh f16 [B*H,S,64], scaled by 1/sqrt(d)
// MODE 1: K -> o16 = Kh f16 [B*H,S,64]; o32 = Kh f32 (d_out section)
// MODE 2: V -> o16 = Vt f16 [B*H,64,S]; o32 = Vh f32 [B*H,S,64] (d_out section)
// MODE 3: O -> o32 = out f32 [ROWS,EMB]
// ---------------------------------------------------------------------------
template <int MODE>
__global__ __launch_bounds__(256) void gemm16(
    const _Float16* __restrict__ X, const _Float16* __restrict__ W,
    const float* __restrict__ bias, _Float16* __restrict__ o16,
    float* __restrict__ o32) {
  const int lane = threadIdx.x & 31;
  const int wv   = threadIdx.x >> 5;
  const int wid  = blockIdx.x * 8 + wv;     // 4096 waves
  const int mp   = wid >> 4;                // 0..255 (32-row strip)
  const int nb   = (wid & 15) << 6;         // 0..960 (N base, 64 wide)
  const int nl   = lane & 15;
  const int hi8  = (lane >> 4) << 3;        // 0 or 8

  const _Float16* arow0 = X + (size_t)(mp * 32 + nl) * EMB;
  const _Float16* arow1 = arow0 + (size_t)16 * EMB;
  const _Float16* brow  = W + (size_t)(nb + nl) * EMB;

  v8f acc[2][4] = {};
  for (int kk = 0; kk < EMB; kk += 32) {
    const int base = kk + hi8;
    __builtin_prefetch(arow0 + base + 128, 0, 0);   // global_prefetch_b8
    __builtin_prefetch(brow + base + 128, 0, 0);
    v16h a0 = ld_frag(arow0, base);
    v16h a1 = ld_frag(arow1, base);
#pragma unroll
    for (int j = 0; j < 4; ++j) {
      v16h b = ld_frag(brow + (size_t)j * 16 * EMB, base);
      acc[0][j] = wmma_f16(a0, b, acc[0][j]);
      acc[1][j] = wmma_f16(a1, b, acc[1][j]);
    }
  }

  float bj[4];
#pragma unroll
  for (int j = 0; j < 4; ++j) bj[j] = bias[nb + j * 16 + nl];

#pragma unroll
  for (int u = 0; u < 2; ++u) {
#pragma unroll
    for (int j = 0; j < 4; ++j) {
      const int c = nb + j * 16 + nl;       // output feature
      const int h = c >> 6, dd = c & 63;
#pragma unroll
      for (int i = 0; i < 8; ++i) {
        const int r = mp * 32 + u * 16 + i + hi8;   // global row (b*S + s)
        const float val = acc[u][j][i] + bj[j];
        if (MODE == 0) {
          const int b = r >> 11, s = r & 2047;
          o16[((size_t)(b * HEADS + h) * SEQ + s) * HDIM + dd] =
              (_Float16)(val * 0.125f);     // fold 1/sqrt(64) into Q
        } else if (MODE == 1) {
          const int b = r >> 11, s = r & 2047;
          const size_t idx = ((size_t)(b * HEADS + h) * SEQ + s) * HDIM + dd;
          o16[idx] = (_Float16)val;
          o32[idx] = val;
        } else if (MODE == 2) {
          const int b = r >> 11, s = r & 2047;
          o32[((size_t)(b * HEADS + h) * SEQ + s) * HDIM + dd] = val;
          o16[((size_t)(b * HEADS + h) * HDIM + dd) * SEQ + s] = (_Float16)val;
        } else {
          o32[(size_t)r * EMB + c] = val;
        }
      }
    }
  }
}

// ---------------------------------------------------------------------------
// Flash attention, one wave per (b,h, 16-query tile), 32 keys per step.
// Softmax reductions are done row-wise after transposing scores through LDS:
// each lane owns half a row (16 contiguous f32), reduces in-lane, and one
// __shfl_xor(16) merges halves. No per-slot ds_bpermute butterflies.
// ---------------------------------------------------------------------------
__global__ __launch_bounds__(128) void attn16(
    const _Float16* __restrict__ Qh,   // [B*H, S, 64] (pre-scaled)
    const _Float16* __restrict__ Kh,   // [B*H, S, 64]
    const _Float16* __restrict__ Vt,   // [B*H, 64, S]
    _Float16* __restrict__ Zh) {       // [ROWS, EMB] f16
  __shared__ __align__(16) float    sS[4][16 * S_STRIDE];  // raw scores
  __shared__ __align__(16) _Float16 sP[4][16 * P_STRIDE];  // exp(P) f16
  __shared__ __align__(16) float    sA[4][16];             // per-row alpha / l

  const int lane = threadIdx.x & 31;
  const int wv   = threadIdx.x >> 5;
  const int wid  = blockIdx.x * 4 + wv;   // 8192 waves
  const int bh   = wid >> 7;              // 0..63
  const int qb   = (wid & 127) << 4;      // query tile base
  const int nl   = lane & 15;
  const int hi8  = (lane >> 4) << 3;      // 0/8  (C-layout half)
  const int rrow = lane & 15;             // row this lane reduces
  const int ch   = (lane >> 4) << 4;      // 0/16 (column half for row reads)

  float* S = sS[wv];
  _Float16* P = sP[wv];
  float* A = sA[wv];

  // Q fragments (16 queries x 64 dims = two 16x32 A-fragments), loaded once
  const _Float16* qrow = Qh + ((size_t)bh * SEQ + qb + nl) * HDIM;
  const v16h qa0 = ld_frag(qrow, hi8);
  const v16h qa1 = ld_frag(qrow, 32 + hi8);

  float m_run = -3.0e38f, l_run = 0.0f;   // running stats for row `rrow`
  v8f o[4] = {};

  for (int kb = 0; kb < SEQ; kb += 32) {
    // ---- scores: two 16x16 tiles -> straight into LDS (slot layout) ----
#pragma unroll
    for (int t = 0; t < 2; ++t) {
      const _Float16* krow = Kh + ((size_t)bh * SEQ + kb + t * 16 + nl) * HDIM;
      v8f s = {};
      s = wmma_f16(qa0, ld_frag(krow, hi8), s);
      s = wmma_f16(qa1, ld_frag(krow, 32 + hi8), s);
#pragma unroll
      for (int i = 0; i < 8; ++i)
        S[(i + hi8) * S_STRIDE + t * 16 + nl] = s[i];
    }
    asm volatile("s_wait_dscnt 0" ::: "memory");

    // ---- row-wise softmax: lane reads 16 contiguous f32 of its row ----
    const float* srow = S + rrow * S_STRIDE + ch;
    float4 r0 = *(const float4*)(srow + 0);
    float4 r1 = *(const float4*)(srow + 4);
    float4 r2 = *(const float4*)(srow + 8);
    float4 r3 = *(const float4*)(srow + 12);
    float v[16] = { r0.x, r0.y, r0.z, r0.w, r1.x, r1.y, r1.z, r1.w,
                    r2.x, r2.y, r2.z, r2.w, r3.x, r3.y, r3.z, r3.w };
    float mx = v[0];
#pragma unroll
    for (int c = 1; c < 16; ++c) mx = fmaxf(mx, v[c]);
    mx = fmaxf(mx, __shfl_xor(mx, 16));         // merge column halves
    const float mnew  = fmaxf(m_run, mx);
    const float alpha = __expf(m_run - mnew);
    m_run = mnew;

    float sum = 0.0f;
    v8h h0, h1;
#pragma unroll
    for (int c = 0; c < 8; ++c) {
      float p = __expf(v[c] - mnew);      sum += p; h0[c] = (_Float16)p;
      float q = __expf(v[c + 8] - mnew);  sum += q; h1[c] = (_Float16)q;
    }
    sum += __shfl_xor(sum, 16);                 // full row sum
    l_run = l_run * alpha + sum;

    // P (f16, row-major) + alpha to LDS
    *(v8h*)(P + rrow * P_STRIDE + ch)     = h0;
    *(v8h*)(P + rrow * P_STRIDE + ch + 8) = h1;
    if (lane < 16) A[rrow] = alpha;
    asm volatile("s_wait_dscnt 0" ::: "memory");

    // ---- rescale O: alpha for slot rows i+hi8 via LDS broadcast ----
    const float4 a0 = *(const float4*)(A + hi8);
    const float4 a1 = *(const float4*)(A + hi8 + 4);
    const float av[8] = { a0.x, a0.y, a0.z, a0.w, a1.x, a1.y, a1.z, a1.w };
#pragma unroll
    for (int j = 0; j < 4; ++j)
#pragma unroll
      for (int i = 0; i < 8; ++i) o[j][i] *= av[i];

    // ---- O += P @ V : A-fragment of P, 4 d-tiles of 16, K=32 keys ----
    const v16h pf = ld_frag(P + nl * P_STRIDE, hi8);
#pragma unroll
    for (int j = 0; j < 4; ++j) {
      const _Float16* vrow = Vt + ((size_t)bh * HDIM + j * 16 + nl) * SEQ + kb;
      o[j] = wmma_f16(pf, ld_frag(vrow, hi8), o[j]);
    }
  }

  // ---- normalize: broadcast per-row l the same way, then store z (f16) ----
  if (lane < 16) A[rrow] = l_run;
  asm volatile("s_wait_dscnt 0" ::: "memory");
  const float4 l0 = *(const float4*)(A + hi8);
  const float4 l1 = *(const float4*)(A + hi8 + 4);
  const float lv[8] = { l0.x, l0.y, l0.z, l0.w, l1.x, l1.y, l1.z, l1.w };

  const int b = bh >> 4, h = bh & 15;
#pragma unroll
  for (int i = 0; i < 8; ++i) {
    const float rinv = 1.0f / lv[i];
    const int s = qb + i + hi8;
#pragma unroll
    for (int j = 0; j < 4; ++j)
      Zh[((size_t)(b * SEQ + s)) * EMB + h * HDIM + j * 16 + nl] =
          (_Float16)(o[j][i] * rinv);
  }
}

// ---------------------------------------------------------------------------
extern "C" void kernel_launch(void* const* d_in, const int* in_sizes, int n_in,
                              void* d_out, int out_size, void* d_ws, size_t ws_size,
                              hipStream_t stream) {
  const float* x    = (const float*)d_in[0];
  const float* Wq_w = (const float*)d_in[1];
  const float* Wq_b = (const float*)d_in[2];
  const float* Wk_w = (const float*)d_in[3];
  const float* Wk_b = (const float*)d_in[4];
  const float* Wv_w = (const float*)d_in[5];
  const float* Wv_b = (const float*)d_in[6];
  const float* Wo_w = (const float*)d_in[7];
  const float* Wo_b = (const float*)d_in[8];

  float* out = (float*)d_out;                       // [4,2048,1024]
  float* KhF = out + (size_t)ROWS * EMB;            // [4,16,2048,64]
  float* VhF = KhF + (size_t)ROWS * EMB;            // [4,16,2048,64]

  char* ws = (char*)d_ws;
  _Float16* xh   = (_Float16*)ws; ws += (size_t)ROWS * EMB * 2;   // 16 MB
  _Float16* wqh  = (_Float16*)ws; ws += (size_t)EMB * EMB * 2;    //  2 MB
  _Float16* wkh  = (_Float16*)ws; ws += (size_t)EMB * EMB * 2;
  _Float16* wvh  = (_Float16*)ws; ws += (size_t)EMB * EMB * 2;
  _Float16* woh  = (_Float16*)ws; ws += (size_t)EMB * EMB * 2;
  _Float16* Qh   = (_Float16*)ws; ws += (size_t)ROWS * EMB * 2;   // 16 MB
  _Float16* Kh16 = (_Float16*)ws; ws += (size_t)ROWS * EMB * 2;
  _Float16* Vt16 = (_Float16*)ws; ws += (size_t)ROWS * EMB * 2;
  _Float16* Zh   = (_Float16*)ws; ws += (size_t)ROWS * EMB * 2;

  const int n4x = ROWS * EMB / 4;
  const int n4w = EMB * EMB / 4;
  cvt_f32_f16<<<(n4x + 255) / 256, 256, 0, stream>>>((const float4*)x, (v4h*)xh, n4x);
  cvt_f32_f16<<<(n4w + 255) / 256, 256, 0, stream>>>((const float4*)Wq_w, (v4h*)wqh, n4w);
  cvt_f32_f16<<<(n4w + 255) / 256, 256, 0, stream>>>((const float4*)Wk_w, (v4h*)wkh, n4w);
  cvt_f32_f16<<<(n4w + 255) / 256, 256, 0, stream>>>((const float4*)Wv_w, (v4h*)wvh, n4w);
  cvt_f32_f16<<<(n4w + 255) / 256, 256, 0, stream>>>((const float4*)Wo_w, (v4h*)woh, n4w);

  // 4096 waves each: (ROWS/32) M-strips x (EMB/64) N-strips
  gemm16<0><<<512, 256, 0, stream>>>(xh, wqh, Wq_b, Qh, nullptr);
  gemm16<1><<<512, 256, 0, stream>>>(xh, wkh, Wk_b, Kh16, KhF);
  gemm16<2><<<512, 256, 0, stream>>>(xh, wvh, Wv_b, Vt16, VhF);

  attn16<<<2048, 128, 0, stream>>>(Qh, Kh16, Vt16, Zh);

  gemm16<3><<<512, 256, 0, stream>>>(Zh, woh, Wo_b, nullptr, out);
}
